// MultiHeadAttention_154618823212
// MI455X (gfx1250) — matrix operand order repaired and checked by reading, not compiled, not run on hardware
//
#include <hip/hip_runtime.h>

typedef __attribute__((ext_vector_type(16))) __bf16 v16bf;
typedef __attribute__((ext_vector_type(8)))  float  v8f;
typedef __attribute__((ext_vector_type(4)))  unsigned int v4u;
typedef __attribute__((ext_vector_type(8)))  int v8i;
typedef __attribute__((ext_vector_type(4)))  int v4i;

union FragB {
    v16bf v;
    uint4 u[2];
    __bf16 h[16];
};

__device__ inline v8f wmma_bf16(v16bf a, v16bf b, v8f c) {
    // D = A(16x32 bf16) * B(32x16 bf16) + C(16x16 f32)
    return __builtin_amdgcn_wmma_f32_16x16x32_bf16(
        /*neg_a=*/false, a, /*neg_b=*/false, b,
        /*c_mod=*/(short)0, c, /*reuse_a=*/false, /*reuse_b=*/false);
}

__device__ inline v8f vzero8() {
    v8f r;
#pragma unroll
    for (int i = 0; i < 8; ++i) r[i] = 0.0f;
    return r;
}

// ---------------------------------------------------------------------------
// TDM: issue a 2D tile load (global -> LDS), bf16 elements (data_size=2B).
// Descriptor bitfields per CDNA5 ISA D# group0/group1 (Ch.10.8).
// tensor_d0/tile_d0 = contiguous (X) dim, tensor_d1/tile_d1 = row (Y) dim,
// stride0 = elements between consecutive rows. All sizes in elements.
// ---------------------------------------------------------------------------
__device__ inline void tdm_load_2d(unsigned lds_off, const void* gptr,
                                   unsigned tensor_d0, unsigned tensor_d1,
                                   unsigned long long stride0,
                                   unsigned tile_d0, unsigned tile_d1) {
    unsigned long long ga = (unsigned long long)(uintptr_t)gptr;
    v4u g0;
    g0[0] = 1u;                                           // count=1 (valid), user mode
    g0[1] = lds_off;                                      // LDS byte address
    g0[2] = (unsigned)(ga & 0xffffffffull);               // global_addr[31:0]
    g0[3] = (unsigned)((ga >> 32) & 0x1ffffffull) |       // global_addr[56:32]
            (2u << 30);                                   // type=2 ("image")
    v8i g1;
    g1[0] = (int)(1u << 16);                              // wg_mask=0, data_size=1 (2B)
    g1[1] = (int)((tensor_d0 & 0xffffu) << 16);           // atomic_addr=0 | dim0[15:0]
    g1[2] = (int)((tensor_d0 >> 16) | ((tensor_d1 & 0xffffu) << 16));
    g1[3] = (int)((tensor_d1 >> 16) | (tile_d0 << 16));   // dim1[31:16] | tile_dim0
    g1[4] = (int)(tile_d1 & 0xffffu);                     // tile_dim1 | tile_dim2=0
    g1[5] = (int)(stride0 & 0xffffffffull);               // dim0_stride[31:0]
    g1[6] = (int)((stride0 >> 32) & 0xffffull);           // dim0_stride[47:32] | s1=0
    g1[7] = 0;
    v4i z;
    z[0] = z[1] = z[2] = z[3] = 0;
#if defined(__clang_major__) && (__clang_major__ >= 23)
    v8i z8;
#pragma unroll
    for (int i = 0; i < 8; ++i) z8[i] = 0;
    __builtin_amdgcn_tensor_load_to_lds(g0, g1, z, z, z8, 0);
#else
    __builtin_amdgcn_tensor_load_to_lds(g0, g1, z, z, 0);
#endif
}

// ---------------------------------------------------------------------------
// fp32 -> bf16 conversion (4 elements/thread)
// ---------------------------------------------------------------------------
__global__ __launch_bounds__(256) void cvt_f32_to_bf16(
    const float* __restrict__ in, __bf16* __restrict__ out, int n4) {
    int i = blockIdx.x * blockDim.x + threadIdx.x;
    if (i >= n4) return;
    float4 f = ((const float4*)in)[i];
    union { __bf16 h[4]; unsigned long long u; } p;
    p.h[0] = (__bf16)f.x; p.h[1] = (__bf16)f.y;
    p.h[2] = (__bf16)f.z; p.h[3] = (__bf16)f.w;
    ((unsigned long long*)out)[i] = p.u;
}

// ---------------------------------------------------------------------------
// bf16 GEMM with TDM double-buffered tile staging.
// C[M,N] = A[M,K] @ B[K,N] (+bias). Block tile 128x128, BK=32, 256 threads
// = 8 waves (4 M x 2 N), each wave computes 32x64 = 2x4 WMMA 16x16x32 tiles.
// Wave 0 drives the Tensor Data Mover; TENSORcnt + barriers synchronize.
// ---------------------------------------------------------------------------
#define BM 128
#define BN 128
#define BK 32

template <bool OUT_BF16>
__global__ __launch_bounds__(256) void gemm_bf16_wmma(
    const __bf16* __restrict__ A, const __bf16* __restrict__ B,
    const float* __restrict__ bias,
    __bf16* __restrict__ Cb, float* __restrict__ Cf,
    int M, int N, int K) {
    __shared__ __bf16 As[2][BM][BK];   // 16 KB, row-major tiles
    __shared__ __bf16 Bs[2][BK][BN];   // 16 KB, row-major tiles

    const int tid  = threadIdx.x;
    const int lane = tid & 31;
    const int wave = tid >> 5;       // 0..7
    const int wm   = wave & 3;       // M-tile group (rows wm*32)
    const int wn   = wave >> 2;      // N-tile group (cols wn*64)
    const int gh   = lane >> 4;      // half-group 0/1
    const int l15  = lane & 15;
    const int blockM = blockIdx.y * BM;
    const int blockN = blockIdx.x * BN;

    const unsigned asBase = (unsigned)(uintptr_t)(&As[0][0][0]);
    const unsigned bsBase = (unsigned)(uintptr_t)(&Bs[0][0][0]);
    const unsigned asBufBytes = BM * BK * 2;
    const unsigned bsBufBytes = BK * BN * 2;

    v8f acc[2][4];
#pragma unroll
    for (int mt = 0; mt < 2; ++mt)
#pragma unroll
        for (int nt = 0; nt < 4; ++nt) acc[mt][nt] = vzero8();

    const int nIter = K / BK;

    // prologue: DMA tile 0 into buffer 0
    if (wave == 0) {
        tdm_load_2d(asBase, A + (size_t)blockM * K, (unsigned)K, (unsigned)M,
                    (unsigned long long)K, BK, BM);
        tdm_load_2d(bsBase, B + blockN, (unsigned)N, (unsigned)K,
                    (unsigned long long)N, BN, BK);
    }

    for (int i = 0; i < nIter; ++i) {
        const int buf = i & 1;
        if (wave == 0) __builtin_amdgcn_s_wait_tensorcnt(0);
        __syncthreads();   // tile i visible to all; previous buffer free

        // DMA tile i+1 into the other buffer while computing tile i
        if (wave == 0 && (i + 1) < nIter) {
            const int k0n = (i + 1) * BK;
            tdm_load_2d(asBase + (unsigned)(1 - buf) * asBufBytes,
                        A + (size_t)blockM * K + k0n, (unsigned)K, (unsigned)M,
                        (unsigned long long)K, BK, BM);
            tdm_load_2d(bsBase + (unsigned)(1 - buf) * bsBufBytes,
                        B + (size_t)k0n * N + blockN, (unsigned)N, (unsigned)K,
                        (unsigned long long)N, BN, BK);
        }

        // --- A fragments: contiguous 16B LDS loads ---
        FragB af[2];
#pragma unroll
        for (int mt = 0; mt < 2; ++mt) {
            int row = wm * 32 + mt * 16 + l15;
            // A layout: e<8 -> K = 8*gh + e ; e>=8 -> K = 16 + 8*gh + (e-8)
            af[mt].u[0] = *(const uint4*)&As[buf][row][gh * 8];
            af[mt].u[1] = *(const uint4*)&As[buf][row][16 + gh * 8];
        }
        // --- B fragments: e<8 -> K = 8*gh + e ; e>=8 -> K = 16 + 8*gh + (e-8), gathered from row-major Bs ---
        FragB bfr[4];
#pragma unroll
        for (int nt = 0; nt < 4; ++nt) {
            int n = wn * 64 + nt * 16 + l15;
#pragma unroll
            for (int e = 0; e < 16; ++e)
                bfr[nt].h[e] = Bs[buf][(e < 8) ? (8 * gh + e) : (16 + 8 * gh + (e - 8))][n];
        }
#pragma unroll
        for (int mt = 0; mt < 2; ++mt)
#pragma unroll
            for (int nt = 0; nt < 4; ++nt)
                acc[mt][nt] = wmma_bf16(af[mt].v, bfr[nt].v, acc[mt][nt]);

        __syncthreads();   // all reads of this buffer done before overwrite
    }

    // --- epilogue: C/D layout lane->(n=l15), vgpr v -> m = v + 8*gh ---
#pragma unroll
    for (int mt = 0; mt < 2; ++mt)
#pragma unroll
        for (int nt = 0; nt < 4; ++nt)
#pragma unroll
            for (int v = 0; v < 8; ++v) {
                int row = blockM + wm * 32 + mt * 16 + v + 8 * gh;
                int col = blockN + wn * 64 + nt * 16 + l15;
                float val = acc[mt][nt][v];
                if constexpr (OUT_BF16) {
                    Cb[(size_t)row * N + col] = (__bf16)val;
                } else {
                    Cf[(size_t)row * N + col] = val + bias[col];
                }
            }
}

// ---------------------------------------------------------------------------
// V transpose: [b,s,h,hd] -> Vt[b,h,hd,s]
// ---------------------------------------------------------------------------
__global__ __launch_bounds__(256) void transpose_v(
    const __bf16* __restrict__ V, __bf16* __restrict__ Vt) {
    int i = blockIdx.x * 256 + threadIdx.x;     // over 4*2048*1024 elements
    int hd = i & 63;
    int h  = (i >> 6) & 15;
    int s  = (i >> 10) & 2047;
    int b  = i >> 21;
    Vt[((size_t)(b * 16 + h) * 64 + hd) * 2048 + s] = V[i];
}

// ---------------------------------------------------------------------------
// Flash attention (causal), one wave per 16 query rows of one (b,h).
// Q,K,ctx: [b, s, h, 64] bf16 ; Vt: [b, h, 64, s] bf16.
// Per 32-key block: 4 WMMA for S = Q K^T, online softmax, P staged via LDS
// (column-major) into A-fragment, 4 WMMA for O += P V.
// ---------------------------------------------------------------------------
__global__ __launch_bounds__(256) void flash_attn_wmma(
    const __bf16* __restrict__ Q, const __bf16* __restrict__ K,
    const __bf16* __restrict__ Vt, __bf16* __restrict__ O) {
    __shared__ __bf16 Pt[8][32][16];   // per-wave P^T (keycol-major), 8 KB

    const int tid  = threadIdx.x;
    const int lane = tid & 31;
    const int wave = tid >> 5;
    const int gh   = lane >> 4;
    const int l15  = lane & 15;

    const int gw = blockIdx.x * 8 + wave;   // 0..8191
    const int qb = gw & 127;                // query block (16 rows each)
    const int h  = (gw >> 7) & 15;
    const int b  = gw >> 11;
    const int q0 = qb * 16;

    const int S = 2048, D = 1024;
    const float scale = 0.125f;             // 1/sqrt(64)

    // Q A-fragments (16x64 -> 2 chunks of 16x32), contiguous per lane
    FragB qf[2];
    {
        const __bf16* qrow = Q + ((size_t)(b * S + q0 + l15) * D + h * 64);
#pragma unroll
        for (int c = 0; c < 2; ++c) {
            qf[c].u[0] = *(const uint4*)(qrow + c * 32 + gh * 16);
            qf[c].u[1] = *(const uint4*)(qrow + c * 32 + gh * 16 + 8);
        }
    }

    v8f o[4];
#pragma unroll
    for (int nt = 0; nt < 4; ++nt) o[nt] = vzero8();
    v8f m_run, l_run;
#pragma unroll
    for (int v = 0; v < 8; ++v) { m_run[v] = -1e30f; l_run[v] = 0.0f; }

    const int nkb = (q0 + 15) / 32 + 1;     // causal: key blocks of 32
    for (int kb = 0; kb < nkb; ++kb) {
        const int k0 = kb * 32;

        // prefetch next key block (K rows + Vt rows)
        if (kb + 1 < nkb) {
            __builtin_prefetch(
                K + ((size_t)(b * S + k0 + 32 + l15) * D + h * 64), 0, 3);
            __builtin_prefetch(
                Vt + ((size_t)((b * 16 + h) * 64 + l15)) * 2048 + k0 + 32, 0, 3);
        }

        // ---- S = Q K^T (two 16x16 key tiles) ----
        v8f s0 = vzero8(), s1 = vzero8();
#pragma unroll
        for (int c = 0; c < 2; ++c) {
            FragB kf0, kf1;
            const __bf16* kr0 =
                K + ((size_t)(b * S + k0 + l15) * D + h * 64 + c * 32 + gh * 16);
            kf0.u[0] = *(const uint4*)(kr0);
            kf0.u[1] = *(const uint4*)(kr0 + 8);
            const __bf16* kr1 =
                K + ((size_t)(b * S + k0 + 16 + l15) * D + h * 64 + c * 32 + gh * 16);
            kf1.u[0] = *(const uint4*)(kr1);
            kf1.u[1] = *(const uint4*)(kr1 + 8);
            s0 = wmma_bf16(qf[c].v, kf0.v, s0);
            s1 = wmma_bf16(qf[c].v, kf1.v, s1);
        }

        // ---- scale + causal mask + row max ----
        v8f rmax;
#pragma unroll
        for (int v = 0; v < 8; ++v) {
            int qrow = q0 + v + 8 * gh;
            float x0 = s0[v] * scale;
            float x1 = s1[v] * scale;
            if (k0 + l15 > qrow)      x0 = -1e30f;
            if (k0 + 16 + l15 > qrow) x1 = -1e30f;
            s0[v] = x0; s1[v] = x1;
            rmax[v] = fmaxf(x0, x1);
        }
#pragma unroll
        for (int off = 1; off < 16; off <<= 1)
#pragma unroll
            for (int v = 0; v < 8; ++v)
                rmax[v] = fmaxf(rmax[v], __shfl_xor(rmax[v], off, 32));

        // ---- online softmax update ----
        v8f alpha, rsum;
#pragma unroll
        for (int v = 0; v < 8; ++v) {
            float mnew = fmaxf(m_run[v], rmax[v]);
            alpha[v]   = __expf(m_run[v] - mnew);
            m_run[v]   = mnew;
            float p0 = __expf(s0[v] - mnew);
            float p1 = __expf(s1[v] - mnew);
            s0[v] = p0; s1[v] = p1;
            rsum[v] = p0 + p1;
        }
#pragma unroll
        for (int off = 1; off < 16; off <<= 1)
#pragma unroll
            for (int v = 0; v < 8; ++v)
                rsum[v] += __shfl_xor(rsum[v], off, 32);
#pragma unroll
        for (int v = 0; v < 8; ++v) l_run[v] = l_run[v] * alpha[v] + rsum[v];
#pragma unroll
        for (int nt = 0; nt < 4; ++nt)
#pragma unroll
            for (int v = 0; v < 8; ++v) o[nt][v] *= alpha[v];

        // ---- P (D-layout) -> LDS column-major -> A-fragment ----
        {
            union { __bf16 h8[8]; uint4 u; } pk;
#pragma unroll
            for (int v = 0; v < 8; ++v) pk.h8[v] = (__bf16)s0[v];
            *(uint4*)&Pt[wave][l15][8 * gh] = pk.u;
#pragma unroll
            for (int v = 0; v < 8; ++v) pk.h8[v] = (__bf16)s1[v];
            *(uint4*)&Pt[wave][16 + l15][8 * gh] = pk.u;
        }
        FragB pf;
#pragma unroll
        for (int e = 0; e < 16; ++e) {
            int kk = e + gh * 16;
            pf.h[e] = Pt[wave][kk][l15];   // A layout: m = l15, K = kk
        }

        // ---- O += P V : B-fragments contiguous from Vt ----
#pragma unroll
        for (int nt = 0; nt < 4; ++nt) {
            FragB vf;
            const __bf16* vr =
                Vt + ((size_t)((b * 16 + h) * 64 + nt * 16 + l15)) * 2048 +
                k0 + gh * 16;
            vf.u[0] = *(const uint4*)(vr);
            vf.u[1] = *(const uint4*)(vr + 8);
            o[nt] = wmma_bf16(pf.v, vf.v, o[nt]);
        }
    }

    // ---- finalize: O /= l, store ctx[b, q, h, hd] ----
#pragma unroll
    for (int nt = 0; nt < 4; ++nt)
#pragma unroll
        for (int v = 0; v < 8; ++v) {
            float val = o[nt][v] / l_run[v];
            int row = q0 + v + 8 * gh;
            O[(size_t)(b * S + row) * D + h * 64 + nt * 16 + l15] = (__bf16)val;
        }
}

// ---------------------------------------------------------------------------
// Host launcher
// ---------------------------------------------------------------------------
extern "C" void kernel_launch(void* const* d_in, const int* in_sizes, int n_in,
                              void* d_out, int out_size, void* d_ws, size_t ws_size,
                              hipStream_t stream) {
    const float* x  = (const float*)d_in[0];   // [4,2048,1024]
    const float* Wq = (const float*)d_in[1];   // [1024,1024]
    const float* Wk = (const float*)d_in[2];
    const float* Wv = (const float*)d_in[3];
    const float* Wo = (const float*)d_in[4];
    const float* bo = (const float*)d_in[5];   // [1024]
    float* out = (float*)d_out;

    const int Mrows = 4 * 2048;                // 8192
    const int Dm = 1024;
    const size_t XB = (size_t)Mrows * Dm * sizeof(__bf16);   // 16 MiB
    const size_t WB = (size_t)Dm * Dm * sizeof(__bf16);      // 2 MiB

    char* p = (char*)d_ws;
    __bf16* xb   = (__bf16*)p; p += XB;
    __bf16* wqb  = (__bf16*)p; p += WB;
    __bf16* wkb  = (__bf16*)p; p += WB;
    __bf16* wvb  = (__bf16*)p; p += WB;
    __bf16* wob  = (__bf16*)p; p += WB;
    __bf16* qb   = (__bf16*)p; p += XB;
    __bf16* kbuf = (__bf16*)p; p += XB;
    __bf16* vbuf = (__bf16*)p; p += XB;
    __bf16* vtb  = (__bf16*)p; p += XB;
    __bf16* ctxb = (__bf16*)p; p += XB;

    // fp32 -> bf16
    {
        int n4 = Mrows * Dm / 4;               // x
        cvt_f32_to_bf16<<<n4 / 256, 256, 0, stream>>>(x, xb, n4);
        int w4 = Dm * Dm / 4;
        cvt_f32_to_bf16<<<w4 / 256, 256, 0, stream>>>(Wq, wqb, w4);
        cvt_f32_to_bf16<<<w4 / 256, 256, 0, stream>>>(Wk, wkb, w4);
        cvt_f32_to_bf16<<<w4 / 256, 256, 0, stream>>>(Wv, wvb, w4);
        cvt_f32_to_bf16<<<w4 / 256, 256, 0, stream>>>(Wo, wob, w4);
    }

    dim3 gemm_grid(Dm / BN, Mrows / BM);       // (8, 64)
    // Q / K / V projections
    gemm_bf16_wmma<true><<<gemm_grid, 256, 0, stream>>>(xb, wqb, nullptr, qb, nullptr, Mrows, Dm, Dm);
    gemm_bf16_wmma<true><<<gemm_grid, 256, 0, stream>>>(xb, wkb, nullptr, kbuf, nullptr, Mrows, Dm, Dm);
    gemm_bf16_wmma<true><<<gemm_grid, 256, 0, stream>>>(xb, wvb, nullptr, vbuf, nullptr, Mrows, Dm, Dm);

    // V -> Vt [b,h,hd,s]
    transpose_v<<<(Mrows * Dm) / 256, 256, 0, stream>>>(vbuf, vtb);

    // causal attention: 8192 waves = 1024 blocks x 8 waves
    flash_attn_wmma<<<1024, 256, 0, stream>>>(qb, kbuf, vtb, ctxb);

    // output projection + bias (fp32 out)
    gemm_bf16_wmma<false><<<gemm_grid, 256, 0, stream>>>(ctxb, wob, bo, nullptr, out, Mrows, Dm, Dm);
}
